// GCN_56822417326210
// MI455X (gfx1250) — compile-verified, matrix operand order checked by reading, not run on hardware
//
#include <hip/hip_runtime.h>

// ---------------------------------------------------------------------------
// GCN forward for MI455X (gfx1250, wave32).
//   h   = relu( spmm(A, x@W1) + b1 )
//   out =       spmm(A, h@W2) + b2
// Dense GEMMs: v_wmma_f32_16x16x32_bf16. Block computes a 128x64 output tile:
// the 64-wide packed B panel is staged in LDS once (shared by 8 waves), each
// wave runs a fully-unrolled K-loop with 4 accumulators (A fragment reused
// across 4 WMMAs, B via ds_load).
// Sparse SPMM: one wave per edge, float4 gathers + f32 global atomics; both
// gathered matrices are L2-resident (102 MB / 25 MB < 192 MB L2).
// ---------------------------------------------------------------------------

typedef __attribute__((ext_vector_type(16))) __bf16 v16bf;
typedef __attribute__((ext_vector_type(8)))  __bf16 v8bf;
typedef __attribute__((ext_vector_type(4)))  __bf16 v4bf;
typedef __attribute__((ext_vector_type(8)))  float  v8f;

#define N_NODES 100000
#define E_EDGES 3200000
#define IN_DIM  128
#define HID     256
#define OUT_DIM 64

// ---------------- elementwise helpers (float4-vectorized) ----------------

__global__ void f32_to_bf16_x4_kernel(const float4* __restrict__ in,
                                      v4bf* __restrict__ out, int n4) {
  int i = blockIdx.x * blockDim.x + threadIdx.x;
  if (i < n4) {
    float4 v = in[i];
    v4bf o;
    o[0] = (__bf16)v.x; o[1] = (__bf16)v.y;
    o[2] = (__bf16)v.z; o[3] = (__bf16)v.w;
    out[i] = o;
  }
}

__global__ void zero_f32_kernel(float4* __restrict__ p, int n4) {
  int i = blockIdx.x * blockDim.x + threadIdx.x;
  if (i < n4) p[i] = make_float4(0.f, 0.f, 0.f, 0.f);
}

// out[i] = bias[i & mask]  (F power of two)
__global__ void init_bias_kernel(float* __restrict__ out,
                                 const float* __restrict__ bias,
                                 int mask, int total) {
  int i = blockIdx.x * blockDim.x + threadIdx.x;
  if (i < total) out[i] = bias[i & mask];
}

// h = relu(acc + b1) -> bf16, 4 elements per thread
__global__ void bias_relu_bf16_x4_kernel(const float4* __restrict__ acc,
                                         const float* __restrict__ bias,
                                         v4bf* __restrict__ out,
                                         int mask, int n4) {
  int i = blockIdx.x * blockDim.x + threadIdx.x;
  if (i < n4) {
    float4 a = acc[i];
    const float4 b = *(const float4*)(bias + ((i << 2) & mask));
    float4 v = make_float4(a.x + b.x, a.y + b.y, a.z + b.z, a.w + b.w);
    v4bf o;
    o[0] = (__bf16)(v.x > 0.f ? v.x : 0.f);
    o[1] = (__bf16)(v.y > 0.f ? v.y : 0.f);
    o[2] = (__bf16)(v.z > 0.f ? v.z : 0.f);
    o[3] = (__bf16)(v.w > 0.f ? v.w : 0.f);
    out[i] = o;
  }
}

// ---------------- B-operand prepack ----------------
// Pack W[K][Ncols] (row-major f32) into the wave32 WMMA B layout, one 16-elt
// (32 B) contiguous chunk per lane per (kstep, nTile):
//   lane<16 : col = nTile*16 + lane,    K = k0 + {0..7, 16..23}
//   lane>=16: col = nTile*16 + lane-16, K = k0 + {8..15, 24..31}
__global__ void prepack_B_kernel(const float* __restrict__ W,
                                 __bf16* __restrict__ Bp,
                                 int K, int Ncols) {
  int p = blockIdx.x * blockDim.x + threadIdx.x;
  int total = K * Ncols;
  if (p >= total) return;
  int j      = p & 15;
  int lane   = (p >> 4) & 31;
  int tile   = p >> 9;              // (kstep, nTile)
  int nTiles = Ncols >> 4;
  int kstep  = tile / nTiles;
  int nTile  = tile - kstep * nTiles;
  int half   = lane >> 4;
  int col    = nTile * 16 + (lane & 15);
  int k      = kstep * 32 + (j & 7) + 8 * half + ((j >> 3) << 4);
  Bp[p] = (__bf16)W[k * Ncols + col];
}

// ---------------- WMMA GEMM: C[M,NCOLS] = A[M,K] * Bpacked ----------------
// Block (256 thr = 8 waves) computes 128 rows x 64 cols. The 64-col packed B
// panel (K*64 bf16) is staged in LDS once, then each wave runs the unrolled
// K-loop: 1 A fragment (2 global b128 loads) feeding 4 WMMAs whose B comes
// from ds_load_b128.
template <int K, int NCOLS>
__global__ void wmma_gemm_bf16_kernel(const __bf16* __restrict__ A,
                                      const __bf16* __restrict__ Bp,
                                      float* __restrict__ C, int M) {
  constexpr int nTiles  = NCOLS / 16;
  constexpr int nGroups = nTiles / 4;         // 64-col groups
  constexpr int KSTEPS  = K / 32;
  constexpr int ELTS_PER_KS = 4 * 32 * 16;    // 2048 bf16 per K-step panel

  __shared__ __bf16 Blds[K * 64];

  int tid    = threadIdx.x;
  int waveId = tid >> 5;
  int lane   = tid & 31;
  int mBlock = blockIdx.x / nGroups;
  int group  = blockIdx.x - mBlock * nGroups;

  // cooperative stage: group's packed B panel -> LDS (contiguous per K-step)
#pragma unroll
  for (int ks = 0; ks < KSTEPS; ++ks) {
    const __bf16* srcp =
        Bp + ((size_t)(ks * nTiles + group * 4) * 32) * 16 + tid * 8;
    *(v8bf*)&Blds[ks * ELTS_PER_KS + tid * 8] = *(const v8bf*)srcp;
  }
  __syncthreads();

  int mTiles = M >> 4;
  int mTile  = mBlock * 8 + waveId;
  if (mTile >= mTiles) return;               // wave-uniform: EXEC all-ones below

  int row = mTile * 16 + (lane & 15);
  int kb  = (lane >> 4) << 3;                // +8 for upper half-wave
  const __bf16* Arow = A + (long long)row * K;

  v8f acc0 = {}, acc1 = {}, acc2 = {}, acc3 = {};
#pragma unroll
  for (int ks = 0; ks < KSTEPS; ++ks) {
    int k0 = ks << 5;
    union { v16bf v; v8bf h[2]; } a;
    a.h[0] = *(const v8bf*)(Arow + k0 + kb);
    a.h[1] = *(const v8bf*)(Arow + k0 + kb + 16);
    const __bf16* Bl = &Blds[ks * ELTS_PER_KS + lane * 16];
    v16bf b0 = *(const v16bf*)(Bl);
    v16bf b1 = *(const v16bf*)(Bl + 512);
    v16bf b2 = *(const v16bf*)(Bl + 1024);
    v16bf b3 = *(const v16bf*)(Bl + 1536);
    acc0 = __builtin_amdgcn_wmma_f32_16x16x32_bf16(false, a.v, false, b0,
                                                   (short)0, acc0, false, false);
    acc1 = __builtin_amdgcn_wmma_f32_16x16x32_bf16(false, a.v, false, b1,
                                                   (short)0, acc1, false, false);
    acc2 = __builtin_amdgcn_wmma_f32_16x16x32_bf16(false, a.v, false, b2,
                                                   (short)0, acc2, false, false);
    acc3 = __builtin_amdgcn_wmma_f32_16x16x32_bf16(false, a.v, false, b3,
                                                   (short)0, acc3, false, false);
  }

  // C/D layout: VGPR r -> M = mTile*16 + r + 8*(lane>=16), N = col
  int rowBase = mTile * 16 + ((lane >> 4) << 3);
  int colBase = group * 64 + (lane & 15);
  float* Cp = C + (long long)rowBase * NCOLS + colBase;
#pragma unroll
  for (int r = 0; r < 8; ++r) {
    float* rp = Cp + (long long)r * NCOLS;
    rp[0]  = acc0[r];
    rp[16] = acc1[r];
    rp[32] = acc2[r];
    rp[48] = acc3[r];
  }
}

// ---------------- SPMM: Hout[dst[e]] += w[e] * Hin[src[e]] ----------------
// One wave per edge; lanes cover the feature dimension with float4 gathers and
// f32 global atomics. Edge id forced uniform so metadata is fetched once.
template <int F>
__global__ void spmm_edge_kernel(const int* __restrict__ dst,
                                 const int* __restrict__ src,
                                 const float* __restrict__ ew,
                                 const float* __restrict__ Hin,
                                 float* __restrict__ Hout, int E) {
  int wid = (int)((blockIdx.x * blockDim.x + threadIdx.x) >> 5);
  int lane = threadIdx.x & 31;
  if (wid >= E) return;                    // wave-uniform
  int e = __builtin_amdgcn_readfirstlane(wid);

  float w = ew[e];
  long long s = src[e];
  long long d = dst[e];
  const float* hi = Hin + s * F;
  float* ho = Hout + d * F;

  if constexpr (F >= 128) {
#pragma unroll
    for (int b = 0; b < F; b += 128) {
      float4 v = *(const float4*)(hi + b + lane * 4);
      float* o = ho + b + lane * 4;
      atomicAdd(o + 0, w * v.x);
      atomicAdd(o + 1, w * v.y);
      atomicAdd(o + 2, w * v.z);
      atomicAdd(o + 3, w * v.w);
    }
  } else {                                  // F == 64: 2 floats per lane
    float2 v = *(const float2*)(hi + lane * 2);
    float* o = ho + lane * 2;
    atomicAdd(o + 0, w * v.x);
    atomicAdd(o + 1, w * v.y);
  }
}

// ---------------- launch ----------------

extern "C" void kernel_launch(void* const* d_in, const int* in_sizes, int n_in,
                              void* d_out, int out_size, void* d_ws, size_t ws_size,
                              hipStream_t stream) {
  const float* x  = (const float*)d_in[0];
  const int*   ei = (const int*)  d_in[1];   // [2, E]: row0 = dst, row1 = src
  const float* ew = (const float*)d_in[2];
  const float* W1 = (const float*)d_in[3];
  const float* b1 = (const float*)d_in[4];
  const float* W2 = (const float*)d_in[5];
  const float* b2 = (const float*)d_in[6];
  float* out = (float*)d_out;

  const int* e_dst = ei;
  const int* e_src = ei + E_EDGES;

  // workspace layout (bytes), with buffer reuse across phases
  constexpr size_t SZ_XB  = (size_t)N_NODES * IN_DIM * 2;   // x as bf16
  constexpr size_t SZ_W1P = (size_t)IN_DIM * HID * 2;
  constexpr size_t SZ_W2P = (size_t)HID * OUT_DIM * 2;
  constexpr size_t SZ_T1  = (size_t)N_NODES * HID * 4;      // x@W1 (f32)

  constexpr size_t OFF_XB  = 0;
  constexpr size_t OFF_W1P = OFF_XB + SZ_XB;
  constexpr size_t OFF_W2P = OFF_W1P + SZ_W1P;
  constexpr size_t OFF_T1  = OFF_W2P + SZ_W2P;
  constexpr size_t OFF_ACC = OFF_T1 + SZ_T1;                // spmm1 accumulator
  constexpr size_t OFF_HB  = OFF_T1;                        // reuse: T1 dead after spmm1
  constexpr size_t OFF_T2  = OFF_XB;                        // reuse: x_bf16 dead after GEMM1

  char* ws = (char*)d_ws;
  __bf16* XB  = (__bf16*)(ws + OFF_XB);
  __bf16* W1P = (__bf16*)(ws + OFF_W1P);
  __bf16* W2P = (__bf16*)(ws + OFF_W2P);
  float*  T1  = (float*) (ws + OFF_T1);
  float*  ACC = (float*) (ws + OFF_ACC);
  __bf16* HB  = (__bf16*)(ws + OFF_HB);
  float*  T2  = (float*) (ws + OFF_T2);

  const int TPB = 256;

  // 1) x -> bf16 ; prepack W1, W2
  {
    int n4 = N_NODES * IN_DIM / 4;
    f32_to_bf16_x4_kernel<<<(n4 + TPB - 1) / TPB, TPB, 0, stream>>>(
        (const float4*)x, (v4bf*)XB, n4);
    int n1 = IN_DIM * HID;
    prepack_B_kernel<<<(n1 + TPB - 1) / TPB, TPB, 0, stream>>>(W1, W1P, IN_DIM, HID);
    int n2 = HID * OUT_DIM;
    prepack_B_kernel<<<(n2 + TPB - 1) / TPB, TPB, 0, stream>>>(W2, W2P, HID, OUT_DIM);
  }

  // 2) T1 = x @ W1   (ceil(6250/8) mBlocks x 4 col-groups)
  {
    int mBlocks = (N_NODES / 16 + 7) / 8;
    int blocks = mBlocks * (HID / 64);
    wmma_gemm_bf16_kernel<IN_DIM, HID><<<blocks, TPB, 0, stream>>>(XB, W1P, T1,
                                                                   N_NODES);
  }

  // 3) ACC = 0 ; ACC[dst] += w * T1[src]
  {
    int n4 = N_NODES * HID / 4;
    zero_f32_kernel<<<(n4 + TPB - 1) / TPB, TPB, 0, stream>>>((float4*)ACC, n4);
    int blocks = (E_EDGES + 7) / 8;          // 8 waves (edges) per block
    spmm_edge_kernel<HID><<<blocks, TPB, 0, stream>>>(e_dst, e_src, ew, T1, ACC,
                                                      E_EDGES);
  }

  // 4) HB = bf16(relu(ACC + b1))   (reuses T1's buffer)
  {
    int n4 = N_NODES * HID / 4;
    bias_relu_bf16_x4_kernel<<<(n4 + TPB - 1) / TPB, TPB, 0, stream>>>(
        (const float4*)ACC, b1, (v4bf*)HB, HID - 1, n4);
  }

  // 5) T2 = HB @ W2   (ceil(6250/8) mBlocks x 1 col-group)
  {
    int mBlocks = (N_NODES / 16 + 7) / 8;
    int blocks = mBlocks * (OUT_DIM / 64);
    wmma_gemm_bf16_kernel<HID, OUT_DIM><<<blocks, TPB, 0, stream>>>(HB, W2P, T2,
                                                                    N_NODES);
  }

  // 6) out = b2 (broadcast) ; out[dst] += w * T2[src]
  {
    int n = N_NODES * OUT_DIM;
    init_bias_kernel<<<(n + TPB - 1) / TPB, TPB, 0, stream>>>(out, b2,
                                                              OUT_DIM - 1, n);
    int blocks = (E_EDGES + 7) / 8;
    spmm_edge_kernel<OUT_DIM><<<blocks, TPB, 0, stream>>>(e_dst, e_src, ew, T2,
                                                          out, E_EDGES);
  }
}
